// qnn_11192684773473
// MI455X (gfx1250) — matrix operand order repaired and checked by reading, not compiled, not run on hardware
//
#include <hip/hip_runtime.h>

typedef __attribute__((ext_vector_type(16))) _Float16 v16h;
typedef __attribute__((ext_vector_type(8)))  float    v8f;

#define BS     128
#define NR     3
#define C_DATA 64
#define C_GT   3
#define C_IN   67          // 64 data + 3 gt channels
#define C_HID  256
#define HW     1024        // 32*32
#define KPAD   96          // K padded to 3 x 32 for f16 WMMA
#define ROWH   104         // LDS row length in halfs (bank-conflict pad)
#define PBLK   64          // pixels per workgroup
#define NWAVE  4
#define NTHR   (NWAVE * 32)

union FragU {
    struct { uint4 a, b; } q;
    v16h h;
};

// -----------------------------------------------------------------------------
// Kernel 1: pack W1 (f32 [256][67]) into f16 WMMA A-fragment order, K zero-
// padded to 96.  Layout: fragment(mt, ks) = 32 lanes x 16 halfs contiguous.
// A 16x32 layout used here, contiguous k (B matches): lane = M row (lane&15); half h -> K offset
//   k = ks*32 + (lane>>4)*16 + h
// -----------------------------------------------------------------------------
__global__ void pack_w1_kernel(const float* __restrict__ W1,
                               _Float16* __restrict__ out)
{
    int tid = blockIdx.x * blockDim.x + threadIdx.x;
    if (tid >= 16 * 3 * 32 * 16) return;
    int h      = tid & 15;
    int lane   = (tid >> 4) & 31;
    int t2     = tid >> 9;
    int ks     = t2 % 3;
    int mt     = t2 / 3;
    int m      = mt * 16 + (lane & 15);
    int laneHi = lane >> 4;
    int k      = ks * 32 + laneHi * 16 + h;
    float v    = (k < C_IN) ? W1[m * C_IN + k] : 0.0f;
    out[tid]   = (_Float16)v;
}

// -----------------------------------------------------------------------------
// Kernel 2: fused  relu(W1.x+b1) -> sigmoid(W3.r+b3) -> ref-normalize -> z
// One workgroup = 1 batch item x 64 pixels; 4 waves, each wave owns 16 pixels.
// GEMM orientation: M = hidden(256), N = pixels(16/wave), K = 96.
// -----------------------------------------------------------------------------
__global__ void __launch_bounds__(NTHR)
fused_attn_kernel(const float* __restrict__ data,
                  const float* __restrict__ gt,
                  const _Float16* __restrict__ w1p,   // packed A fragments
                  const float* __restrict__ b1,
                  const float* __restrict__ W3,
                  const float* __restrict__ b3p,
                  float* __restrict__ out)
{
    __shared__ __align__(16) _Float16 xlds[NR][PBLK][ROWH]; // ~39 KB
    __shared__ float slds[NR][PBLK];

    const int tid    = threadIdx.x;
    const int lane   = tid & 31;
    const int wv     = tid >> 5;
    const int laneLo = lane & 15;
    const int laneHi = lane >> 4;
    const int b      = blockIdx.x >> 4;
    const int pbase  = (blockIdx.x & 15) * PBLK;

    // ---------------- stage X tiles: f32 global -> f16 LDS [pixel][K] --------
    {
        const int p4 = (tid & 15) * 4;   // 4 consecutive pixels per thread
        const int cq = tid >> 4;         // 0..7
        for (int n = 0; n < NR; ++n) {
            const float* dsrc =
                data + (size_t)(b * NR + n) * C_DATA * HW + pbase;
            #pragma unroll
            for (int pass = 0; pass < 8; ++pass) {
                int c = pass * 8 + cq;
                float4 v = *(const float4*)(dsrc + (size_t)c * HW + p4);
                xlds[n][p4 + 0][c] = (_Float16)v.x;
                xlds[n][p4 + 1][c] = (_Float16)v.y;
                xlds[n][p4 + 2][c] = (_Float16)v.z;
                xlds[n][p4 + 3][c] = (_Float16)v.w;
            }
            if (tid < 48) {              // gt channels 64..66 (broadcast over n)
                int cg = tid >> 4;       // 0..2
                float4 v = *(const float4*)(gt + ((size_t)b * C_GT + cg) * HW
                                            + pbase + p4);
                xlds[n][p4 + 0][C_DATA + cg] = (_Float16)v.x;
                xlds[n][p4 + 1][C_DATA + cg] = (_Float16)v.y;
                xlds[n][p4 + 2][C_DATA + cg] = (_Float16)v.z;
                xlds[n][p4 + 3][C_DATA + cg] = (_Float16)v.w;
            }
            // zero K = 67..103 so padded WMMA lanes can't see NaN garbage
            for (int idx = tid; idx < PBLK * (ROWH - C_IN); idx += NTHR) {
                int p = idx / (ROWH - C_IN);
                int k = C_IN + idx % (ROWH - C_IN);
                xlds[n][p][k] = (_Float16)0.0f;
            }
        }
    }
    __syncthreads();

    // ---------------- GEMM1 + bias + relu + GEMM2 (fused per M-tile) ---------
    float tacc[NR] = {0.0f, 0.0f, 0.0f};
    const int prow = wv * 16 + laneLo;   // this lane's pixel (= WMMA N index)

    for (int mt = 0; mt < 16; ++mt) {
        // A fragments for the 3 K-steps (shared across refs; L0/L2 hits)
        FragU af[3];
        #pragma unroll
        for (int ks = 0; ks < 3; ++ks) {
            const uint4* s = (const uint4*)
                (w1p + (size_t)((mt * 3 + ks) * 32 + lane) * 16);
            af[ks].q.a = s[0];
            af[ks].q.b = s[1];
        }
        // bias / W3 slices for this lane's 8 hidden rows: m = mt*16+laneHi*8+r
        const int mb = mt * 16 + laneHi * 8;
        float4 b1lo = *(const float4*)(b1 + mb);
        float4 b1hi = *(const float4*)(b1 + mb + 4);
        float4 w3lo = *(const float4*)(W3 + mb);
        float4 w3hi = *(const float4*)(W3 + mb + 4);
        const float b1v[8] = {b1lo.x, b1lo.y, b1lo.z, b1lo.w,
                              b1hi.x, b1hi.y, b1hi.z, b1hi.w};
        const float w3v[8] = {w3lo.x, w3lo.y, w3lo.z, w3lo.w,
                              w3hi.x, w3hi.y, w3hi.z, w3hi.w};

        #pragma unroll
        for (int n = 0; n < NR; ++n) {
            v8f acc = {};
            #pragma unroll
            for (int ks = 0; ks < 3; ++ks) {
                FragU bf;  // B 32x16: lanes 0-15 K=base..base+15, 16-31 +16
                const uint4* s = (const uint4*)
                    (&xlds[n][prow][ks * 32 + laneHi * 16]);
                bf.q.a = s[0];
                bf.q.b = s[1];
                acc = __builtin_amdgcn_wmma_f32_16x16x32_f16(
                    false, af[ks].h, false, bf.h, (short)0, acc, false, false);
            }
            float tp = 0.0f;
            #pragma unroll
            for (int r = 0; r < 8; ++r) {
                float hh = acc[r] + b1v[r];
                hh = fmaxf(hh, 0.0f);
                tp = fmaf(w3v[r], hh, tp);
            }
            tacc[n] += tp;
        }
    }

    // ---------------- sigmoid + stash per-ref scores -------------------------
    const float b3 = b3p[0];
    #pragma unroll
    for (int n = 0; n < NR; ++n) {
        float tfull = tacc[n] + __shfl_xor(tacc[n], 16, 32);
        if (laneHi == 0) {
            float s = 1.0f / (1.0f + __expf(-(tfull + b3)));
            slds[n][wv * 16 + laneLo] = s;
        }
    }
    __syncthreads();

    // ---------------- normalize over refs, write w, weighted-sum z -----------
    const size_t ZSZ = (size_t)BS * C_DATA * HW;
    const int p  = tid & 63;
    const int ch = tid >> 6;             // 0 or 1
    float s0 = slds[0][p], s1 = slds[1][p], s2 = slds[2][p];
    float inv = 1.0f / (s0 + s1 + s2);
    float w0 = s0 * inv, w1 = s1 * inv, w2 = s2 * inv;

    if (tid < PBLK) {
        float* wout = out + ZSZ + (size_t)b * NR * HW + pbase + p;
        wout[0 * HW] = w0;
        wout[1 * HW] = w1;
        wout[2 * HW] = w2;
    }

    const float* d0 = data + (size_t)(b * NR + 0) * C_DATA * HW + pbase + p;
    const float* d1 = data + (size_t)(b * NR + 1) * C_DATA * HW + pbase + p;
    const float* d2 = data + (size_t)(b * NR + 2) * C_DATA * HW + pbase + p;
    float* zout = out + (size_t)b * C_DATA * HW + pbase + p;

    for (int it = 0; it < 32; ++it) {
        size_t off = (size_t)(it * 2 + ch) * HW;
        zout[off] = d0[off] * w0 + d1[off] * w1 + d2[off] * w2;
    }
}

// -----------------------------------------------------------------------------
extern "C" void kernel_launch(void* const* d_in, const int* in_sizes, int n_in,
                              void* d_out, int out_size, void* d_ws,
                              size_t ws_size, hipStream_t stream)
{
    const float* data = (const float*)d_in[0];
    const float* gt   = (const float*)d_in[1];
    const float* W1   = (const float*)d_in[2];
    const float* b1   = (const float*)d_in[3];
    const float* W3   = (const float*)d_in[4];
    const float* b3   = (const float*)d_in[5];
    float* out        = (float*)d_out;
    _Float16* w1p     = (_Float16*)d_ws;      // 16*3*32*16 halfs = 48 KB

    pack_w1_kernel<<<(16 * 3 * 32 * 16 + 255) / 256, 256, 0, stream>>>(W1, w1p);

    const int nblocks = BS * (HW / PBLK);     // 128 * 16 = 2048
    fused_attn_kernel<<<nblocks, NTHR, 0, stream>>>(data, gt, w1p, b1, W3, b3,
                                                    out);
}
